// GNNNodeEncoder_16965120819430
// MI455X (gfx1250) — compile-verified, matrix operand order checked by reading, not run on hardware
//
#include <hip/hip_runtime.h>

// ---------------------------------------------------------------------------
// GNN node encoder for MI455X (gfx1250, wave32, WMMA).
//   h0 = relu([an/10, pos] @ W_in + b_in)
//   h1 = relu(mean_agg(h0) @ W1_l + b1 + h0 @ W1_r)
//   h2 = relu(mean_agg(h1) @ W2_l + b2 + h1 @ W2_r)
//   ns = h2 @ W_out + b_out ; out = (mean(ns, axis=0), ns)
// GEMMs: v_wmma_f32_16x16x32_f16, C-resident (8 N-tile accumulators/wave),
// A streamed from global f32 (cvt on the fly), B from LDS-staged f16 W^T.
// ---------------------------------------------------------------------------

typedef __attribute__((ext_vector_type(16))) _Float16 v16h;
typedef __attribute__((ext_vector_type(8)))  _Float16 half8;
typedef __attribute__((ext_vector_type(8)))  float    v8f;

#define HID 128

// ---- input projection: x=[an/10, pos] (N,4) @ W_in (4,128) + b, relu -------
__global__ void inproj_kernel(const float* __restrict__ pos,
                              const int*   __restrict__ an,
                              const float* __restrict__ W,   // [4,128] row-major
                              const float* __restrict__ b,
                              float* __restrict__ h, int n) {
  int t = blockIdx.x * 256 + threadIdx.x;
  if (t >= n * HID) return;
  int node = t >> 7, c = t & 127;
  float z  = (float)an[node] * 0.1f;
  float px = pos[node * 3 + 0];
  float py = pos[node * 3 + 1];
  float pz = pos[node * 3 + 2];
  float v = b[c];
  v = fmaf(z,  W[0 * HID + c], v);
  v = fmaf(px, W[1 * HID + c], v);
  v = fmaf(py, W[2 * HID + c], v);
  v = fmaf(pz, W[3 * HID + c], v);
  h[t] = fmaxf(v, 0.0f);
}

// ---- one-time weight prep: W f32 [k][n] -> WT f16 [n][k] -------------------
__global__ void wprep_kernel(const float* __restrict__ W,
                             _Float16* __restrict__ WT) {
  int t = blockIdx.x * 256 + threadIdx.x;
  if (t >= HID * HID) return;
  int k = t & 127, n = t >> 7;             // contiguous writes over k
  WT[t] = (_Float16)W[k * HID + n];
}

// ---- zero scratch (s + cnt) ------------------------------------------------
__global__ void zero_kernel(float* __restrict__ p, int n) {
  int t = blockIdx.x * 256 + threadIdx.x;
  if (t < n) p[t] = 0.0f;
}

// ---- edge scatter: one wave per edge, float4 per lane, f32 atomics --------
__global__ void scatter_kernel(const float* __restrict__ h,
                               const int*   __restrict__ ei,  // [2,E] flat
                               float* __restrict__ s,
                               float* __restrict__ cnt, int nE) {
  int t = blockIdx.x * 256 + threadIdx.x;
  int e = t >> 5;
  if (e >= nE) return;
  int lane = t & 31;
  // prefetch the gather row a few edges ahead (-> global_prefetch_b8)
  int ep = e + 4;
  if (ep < nE) {
    int psrc = ei[ep];
    __builtin_prefetch(h + (size_t)psrc * HID + lane * 4, 0, 1);
  }
  int src = ei[e];
  int dst = ei[nE + e];
  const float4 v = *(const float4*)(h + (size_t)src * HID + lane * 4);
  float* dp = s + (size_t)dst * HID + lane * 4;
  atomicAdd(dp + 0, v.x);
  atomicAdd(dp + 1, v.y);
  atomicAdd(dp + 2, v.z);
  atomicAdd(dp + 3, v.w);
  if (lane == 0) atomicAdd(cnt + dst, 1.0f);
}

// ---- helpers for WMMA fragment construction -------------------------------
__device__ inline half8 cvt8(const float* __restrict__ p, float scale) {
  float4 x = *(const float4*)(p);
  float4 y = *(const float4*)(p + 4);
  half8 r;
  r[0] = (_Float16)(x.x * scale); r[1] = (_Float16)(x.y * scale);
  r[2] = (_Float16)(x.z * scale); r[3] = (_Float16)(x.w * scale);
  r[4] = (_Float16)(y.x * scale); r[5] = (_Float16)(y.y * scale);
  r[6] = (_Float16)(y.z * scale); r[7] = (_Float16)(y.w * scale);
  return r;
}

__device__ inline v16h join16(half8 lo, half8 hi) {
  return __builtin_shufflevector(lo, hi, 0, 1, 2, 3, 4, 5, 6, 7,
                                         8, 9, 10, 11, 12, 13, 14, 15);
}

// ---- fused combine GEMM ----------------------------------------------------
// SAGE=true : out = relu?(agg @ Wl + h @ Wr + bias), agg = s/cnt (0 if cnt==0)
// SAGE=false: out = h @ Wr + bias
// 256 threads = 8 wave32; wave owns a 16-row M tile; 8 resident 16x16 f32
// accumulators; K streamed in 4 steps of v_wmma_f32_16x16x32_f16.
template <bool SAGE, bool RELU>
__global__ __launch_bounds__(256) void gemm_kernel(
    const float*    __restrict__ h,
    const float*    __restrict__ s,
    const float*    __restrict__ cnt,
    const _Float16* __restrict__ WlT,   // [n][k] f16 (agg side)
    const _Float16* __restrict__ WrT,   // [n][k] f16 (h side)
    const float*    __restrict__ bias,  // [128]
    float* __restrict__ out,
    int n_nodes) {
  __shared__ _Float16 wtr[HID * HID];
  __shared__ _Float16 wtl[HID * HID];

  // stage pre-transposed f16 weights with b128 copies (coalesced both sides)
  {
    uint4*       dr = (uint4*)wtr;
    const uint4* sr = (const uint4*)WrT;
    for (int i = threadIdx.x; i < HID * HID / 8; i += 256) dr[i] = sr[i];
    if constexpr (SAGE) {
      uint4*       dl = (uint4*)wtl;
      const uint4* sl = (const uint4*)WlT;
      for (int i = threadIdx.x; i < HID * HID / 8; i += 256) dl[i] = sl[i];
    }
  }
  __syncthreads();

  const int lane = threadIdx.x & 31;
  const int wave = threadIdx.x >> 5;
  const int rowbase = blockIdx.x * 128 + wave * 16;
  if (rowbase >= n_nodes) return;          // N % 16 == 0: active tiles full

  // ISA 7.12.2 16-bit A layout: lane holds row M=lane&15;
  // elem e<8 -> K = kb + lo + e ; elem e>=8 -> K = kb + 16 + lo + (e-8)
  const int arow = rowbase + (lane & 15);
  const int row  = arow < n_nodes ? arow : n_nodes - 1;
  const int lo   = (lane >= 16) ? 8 : 0;

  float inv = 0.0f;
  if constexpr (SAGE) {
    float cv = cnt[row];
    // cnt is an exact small integer; fast v_rcp_f32 is plenty (results are
    // rounded to f16 for the WMMA inputs anyway)
    inv = (cv > 0.0f) ? __builtin_amdgcn_rcpf(cv) : 0.0f;
  }

  const float*    hp  = h + (size_t)row * HID + lo;
  const float*    sp  = SAGE ? (s + (size_t)row * HID + lo) : nullptr;
  // B: lane holds column N=lane&15; one base pointer + immediate DS offsets
  const _Float16* brp = &wtr[(lane & 15) * HID + lo];
  const _Float16* blp = &wtl[(lane & 15) * HID + lo];

  v8f acc[8];
  {
    v8f z = {};
#pragma unroll
    for (int nt = 0; nt < 8; ++nt) acc[nt] = z;
  }

#pragma unroll
  for (int kk = 0; kk < 4; ++kk) {
    const int kb = kk * 32;
    v16h ahf = join16(cvt8(hp + kb, 1.0f), cvt8(hp + kb + 16, 1.0f));
    v16h agf;
    if constexpr (SAGE)
      agf = join16(cvt8(sp + kb, inv), cvt8(sp + kb + 16, inv));
#pragma unroll
    for (int nt = 0; nt < 8; ++nt) {
      const int off = nt * 16 * HID + kb;  // nt*2048 + kb halves
      v16h br = join16(*(const half8*)(brp + off),
                       *(const half8*)(brp + off + 16));
      acc[nt] = __builtin_amdgcn_wmma_f32_16x16x32_f16(
          false, ahf, false, br, (short)0, acc[nt], false, false);
      if constexpr (SAGE) {
        v16h bl = join16(*(const half8*)(blp + off),
                         *(const half8*)(blp + off + 16));
        acc[nt] = __builtin_amdgcn_wmma_f32_16x16x32_f16(
            false, agf, false, bl, (short)0, acc[nt], false, false);
      }
    }
  }

  // C/D layout: VGPR r -> M = r + 8*(lane>=16), N = lane&15
  const int mofs = (lane >= 16) ? 8 : 0;
#pragma unroll
  for (int nt = 0; nt < 8; ++nt) {
    const int ncol = nt * 16 + (lane & 15);
    const float bb = bias[ncol];
#pragma unroll
    for (int r = 0; r < 8; ++r) {
      float v = acc[nt][r] + bb;
      if constexpr (RELU) v = fmaxf(v, 0.0f);
      int orow = rowbase + mofs + r;
      if (orow < n_nodes) out[(size_t)orow * HID + ncol] = v;
    }
  }
}

// ---- column mean: one block per column (N*128 f32 fits in 192MB L2) -------
__global__ void colmean_kernel(const float* __restrict__ ns,
                               float* __restrict__ outm, int n) {
  __shared__ float red[256];
  int col = blockIdx.x;
  float acc = 0.0f;
  for (int r = threadIdx.x; r < n; r += 256) acc += ns[(size_t)r * HID + col];
  red[threadIdx.x] = acc;
  __syncthreads();
  for (int st = 128; st > 0; st >>= 1) {
    if (threadIdx.x < st) red[threadIdx.x] += red[threadIdx.x + st];
    __syncthreads();
  }
  if (threadIdx.x == 0) outm[col] = red[0] / (float)n;
}

// ---------------------------------------------------------------------------
extern "C" void kernel_launch(void* const* d_in, const int* in_sizes, int n_in,
                              void* d_out, int out_size, void* d_ws, size_t ws_size,
                              hipStream_t stream) {
  const float* pos  = (const float*)d_in[0];
  const int*   an   = (const int*)  d_in[1];
  const int*   ei   = (const int*)  d_in[2];
  const float* W_in = (const float*)d_in[3];
  const float* b_in = (const float*)d_in[4];
  const float* W1l  = (const float*)d_in[5];
  const float* b1   = (const float*)d_in[6];
  const float* W1r  = (const float*)d_in[7];
  const float* W2l  = (const float*)d_in[8];
  const float* b2   = (const float*)d_in[9];
  const float* W2r  = (const float*)d_in[10];
  const float* Wout = (const float*)d_in[11];
  const float* bout = (const float*)d_in[12];

  const int n  = in_sizes[0] / 3;   // N nodes
  const int nE = in_sizes[2] / 2;   // E edges

  float* ws   = (float*)d_ws;
  float* h    = ws;                          // [N,128] activations (in-place)
  float* s    = ws + (size_t)n * HID;        // [N,128] neighbor sums
  float* cnt  = s  + (size_t)n * HID;        // [N] counts (contiguous with s)

  // f16 transposed weights after cnt, 256B aligned
  size_t woff = (size_t)n * HID * 2 + n;
  woff = (woff + 63) & ~(size_t)63;          // align to 64 floats = 256B
  _Float16* wtbase = (_Float16*)(ws + woff);
  _Float16* W1lT = wtbase + 0 * HID * HID;
  _Float16* W1rT = wtbase + 1 * HID * HID;
  _Float16* W2lT = wtbase + 2 * HID * HID;
  _Float16* W2rT = wtbase + 3 * HID * HID;
  _Float16* WoT  = wtbase + 4 * HID * HID;

  float* outm = (float*)d_out;               // [128] mean
  float* nst  = (float*)d_out + HID;         // [N,128] node_states

  const int nh       = n * HID;
  const int gemmBlks = (n + 127) / 128;
  const int edgeThr  = nE * 32;
  const int wBlks    = (HID * HID + 255) / 256;

  // one-time weight prep (f32 [k][n] -> f16 [n][k])
  wprep_kernel<<<wBlks, 256, 0, stream>>>(W1l,  W1lT);
  wprep_kernel<<<wBlks, 256, 0, stream>>>(W1r,  W1rT);
  wprep_kernel<<<wBlks, 256, 0, stream>>>(W2l,  W2lT);
  wprep_kernel<<<wBlks, 256, 0, stream>>>(W2r,  W2rT);
  wprep_kernel<<<wBlks, 256, 0, stream>>>(Wout, WoT);

  inproj_kernel<<<(nh + 255) / 256, 256, 0, stream>>>(pos, an, W_in, b_in, h, n);

  // conv1
  zero_kernel<<<(nh + n + 255) / 256, 256, 0, stream>>>(s, nh + n);
  scatter_kernel<<<(edgeThr + 255) / 256, 256, 0, stream>>>(h, ei, s, cnt, nE);
  gemm_kernel<true, true><<<gemmBlks, 256, 0, stream>>>(h, s, cnt, W1lT, W1rT, b1, h, n);

  // conv2
  zero_kernel<<<(nh + n + 255) / 256, 256, 0, stream>>>(s, nh + n);
  scatter_kernel<<<(edgeThr + 255) / 256, 256, 0, stream>>>(h, ei, s, cnt, nE);
  gemm_kernel<true, true><<<gemmBlks, 256, 0, stream>>>(h, s, cnt, W2lT, W2rT, b2, h, n);

  // output projection (no relu) + mean
  gemm_kernel<false, false><<<gemmBlks, 256, 0, stream>>>(h, nullptr, nullptr,
                                                          nullptr, WoT, bout, nst, n);
  colmean_kernel<<<HID, 256, 0, stream>>>(nst, outm, n);
}
